// TurboQuantPINN_78176994721996
// MI455X (gfx1250) — compile-verified, matrix operand order checked by reading, not run on hardware
//
#include <hip/hip_runtime.h>

#define BTOT 131072
#define DIM  36
#define HDIM 256
#define ROWS 64
#define SH   264   // padded half stride for h16/t16 (528B -> bank rotation)
#define SWC  136   // padded half stride for a 128-half weight chunk (272B -> rotation)

typedef _Float16 half_t;
typedef _Float16 v16h __attribute__((ext_vector_type(16)));
typedef _Float16 v8h  __attribute__((ext_vector_type(8)));
typedef float    v8f  __attribute__((ext_vector_type(8)));
typedef unsigned int u32x4 __attribute__((ext_vector_type(4)));
typedef int          i32x4 __attribute__((ext_vector_type(4)));
typedef int          i32x8 __attribute__((ext_vector_type(8)));

// ---- LDS layout (bytes) ----
#define OFF_H32 0                              // 64*256*4  = 65536
#define OFF_H16 65536                          // 64*264*2  = 33792
#define OFF_T16 (65536 + 33792)                // 33792
#define OFF_WB0 (65536 + 33792 + 33792)        // 133120 ; 256*136*2 = 69632
#define OFF_WB1 (OFF_WB0 + 69632)              // 202752 ; 69632
#define OFF_RED (OFF_WB1 + 69632)              // 272384 ; 2*256 floats = 2048
#define OFF_MU  (OFF_RED + 2048)               // 274432
#define OFF_RS  (OFF_MU + 256)                 // 274688
#define SMEM_BYTES (OFF_RS + 256)              // 274944 < 320KB

#if defined(__has_builtin)
#  if __has_builtin(__builtin_amdgcn_tensor_load_to_lds) && __has_builtin(__builtin_amdgcn_s_wait_tensorcnt)
#    define USE_TDM 1
#  endif
#endif
#ifndef USE_TDM
#  define USE_TDM 0
#endif

struct MainArgs {
    const float* x;
    const float* fourier;
    const float* inpw; const float* inpb;
    const float* b1[6]; const float* b2[6];
    const float* ln_g; const float* ln_b;
    const float* em1b; const float* em2w; const float* em2b;
    const float* th1b; const float* th2w; const float* th2b;
    const float* st1b; const float* st2w; const float* st2b;
    const half_t* wt;   // 12*65536 block weights (f16, [n][k]), then 160*256 head weights
    float* out;
};

struct WPack { const float* w[12]; };

// 16x32 f16 fragment load (A layout; B uses same addressing on transposed weights).
__device__ __forceinline__ v16h load_frag(const half_t* base, int row0, int stride,
                                          int kb, int lane) {
    const half_t* p = base + (row0 + (lane & 15)) * stride + kb + (((lane >> 4) & 1) << 3);
    v8h lo = *(const v8h*)p;
    v8h hi = *(const v8h*)(p + 16);
    return __builtin_shufflevector(lo, hi, 0,1,2,3,4,5,6,7,8,9,10,11,12,13,14,15);
}

__device__ __forceinline__ float siluf(float v)    { return v / (1.f + __expf(-v)); }
__device__ __forceinline__ float sigmoidf_(float v){ return 1.f / (1.f + __expf(-v)); }

// ---------------- weight-chunk staging: TDM (async DMA) or manual fallback ----------
// Chunk = rows x 128 halves from a [rows x 256] row-major f16 matrix, written to LDS
// with a 136-half row stride (via TDM LDS padding).
#if USE_TDM
__device__ __forceinline__ void tdm_issue_chunk(const half_t* gsrc, unsigned lds_off,
                                                int rows) {
    unsigned long long ga = (unsigned long long)(size_t)gsrc;
    u32x4 g0;
    g0[0] = 1u;                                            // count=1 (valid user D#)
    g0[1] = lds_off;                                       // lds_addr (bytes)
    g0[2] = (unsigned)(ga & 0xffffffffu);                  // global_addr[31:0]
    g0[3] = (unsigned)((ga >> 32) & 0x1ffffffu) | (2u << 30); // addr[56:32] | type=2
    i32x8 g1;
    g1[0] = (1 << 16)        // data_size = 2 bytes
          | (1 << 20)        // pad_enable
          | (5 << 22)        // pad_interval code 5 = 64 DWORDs (one 128-half row)
          | (3 << 25);       // pad_amount  code 3 = 4 DWORDs (8 halves) -> stride 136
    g1[1] = (128 << 16);                 // tensor_dim0[15:0] = 128
    g1[2] = (rows << 16);                // tensor_dim0[31:16]=0 | tensor_dim1[15:0]=rows
    g1[3] = (128 << 16);                 // tensor_dim1[31:16]=0 | tile_dim0 = 128
    g1[4] = rows;                        // tile_dim1 = rows | tile_dim2 = 0 (2D)
    g1[5] = 256;                         // tensor_dim0_stride = 256 elements
    g1[6] = 0;
    g1[7] = 0;
    i32x4 z4 = {0, 0, 0, 0};
    i32x8 z8 = {0, 0, 0, 0, 0, 0, 0, 0};
    __builtin_amdgcn_tensor_load_to_lds(g0, g1, z4, z4, z8, 0);
}
#endif

__device__ __forceinline__ void issue_stage(const half_t* __restrict__ src, half_t* dst,
                                            unsigned dst_off, int rows, int t) {
#if USE_TDM
    (void)dst;
    if (t < 32) tdm_issue_chunk(src, dst_off, rows);       // wave 0 issues the DMA
#else
    (void)dst_off;
    const int chunks = rows * 16;                          // 16 x 8-half pieces per row
    for (int c = t; c < chunks; c += 256) {
        const int n = c >> 4, j = c & 15;
        *(v8h*)(dst + n * SWC + j * 8) = *(const v8h*)(src + (size_t)n * 256 + j * 8);
    }
#endif
}

__device__ __forceinline__ void stage_wait(int t) {
#if USE_TDM
    if (t < 32) __builtin_amdgcn_s_wait_tensorcnt(0);
#else
    (void)t;
#endif
    __syncthreads();
}

// Partial GEMM over one 128-wide K half: 8 tiles/wave (4 M x 2 N), 4 wmma per tile.
__device__ __forceinline__ void gemm_half8(const half_t* A, const half_t* Bm,
                                           int akb, int wave, int lane, v8f acc[8]) {
    int idx = 0;
#pragma unroll
    for (int mt = 0; mt < 4; ++mt) {
#pragma unroll
        for (int nn = 0; nn < 2; ++nn, ++idx) {
            const int nt = wave * 2 + nn;
#pragma unroll
            for (int kt = 0; kt < 4; ++kt) {
                v16h a = load_frag(A,  mt * 16, SH,  akb + kt * 32, lane);
                v16h b = load_frag(Bm, nt * 16, SWC, kt * 32, lane);
                acc[idx] = __builtin_amdgcn_wmma_f32_16x16x32_f16(false, a, false, b,
                                                                  (short)0, acc[idx],
                                                                  false, false);
            }
        }
    }
}

// ---------------- prep kernels: fp32 [k][n] -> fp16 transposed [n][k] ----------------
__global__ __launch_bounds__(256) void prep_block_w(WPack wp, half_t* dst) {
    const int idx = blockIdx.x * 256 + threadIdx.x;   // 12*65536 total
    const int l = idx >> 16;
    const int r = idx & 65535;
    const int n = r >> 8, k = r & 255;
    dst[idx] = (half_t)wp.w[l][k * 256 + n];
}

__global__ __launch_bounds__(256) void prep_head_w(const float* em1w, const float* th1w,
                                                   const float* st1w, half_t* dst) {
    const int idx = blockIdx.x * 256 + threadIdx.x;   // 160*256 total
    const int n = idx >> 8, k = idx & 255;
    float v;
    if (n < 64)       v = em1w[k * 64 + n];
    else if (n < 128) v = th1w[k * 64 + (n - 64)];
    else              v = st1w[k * 32 + (n - 128)];
    dst[idx] = (half_t)v;
}

// ---------------- fused network kernel: 64 rows per workgroup ----------------
__global__ __launch_bounds__(256) void pinn_fused(MainArgs P) {
    extern __shared__ char smem[];
    float*  h32  = (float*) (smem + OFF_H32);
    half_t* h16  = (half_t*)(smem + OFF_H16);
    half_t* t16  = (half_t*)(smem + OFF_T16);
    half_t* wb[2] = { (half_t*)(smem + OFF_WB0), (half_t*)(smem + OFF_WB1) };
    const unsigned woff[2] = { OFF_WB0, OFF_WB1 };
    float*  reds = (float*) (smem + OFF_RED);
    float*  redq = reds + 256;
    float*  muv  = (float*) (smem + OFF_MU);
    float*  rsv  = (float*) (smem + OFF_RS);

    const int t    = threadIdx.x;
    const int lane = t & 31;
    const int wave = t >> 5;
    const int row0 = blockIdx.x * ROWS;
    const half_t* wh = P.wt + (size_t)12 * 65536;   // head-1 weights

    // Kick off DMA of layer-0 W1 K-half 0 immediately; it overlaps stage 0.
    issue_stage(P.wt, wb[0], woff[0], 256, t);
    int cur = 0;

    // ---------- stage 0: x tile -> LDS; fourier + input layer (VALU, K=36) ----------
    {
        float* xl = (float*)t16;   // reuse t16 region as f32 x-tile
        for (int i = t; i < ROWS * DIM; i += 256)
            xl[i] = P.x[(long)row0 * DIM + i];
        __syncthreads();

        float wcol[DIM], fcol[DIM];
        const int fc = t & 127;
#pragma unroll
        for (int k = 0; k < DIM; ++k) {
            wcol[k] = P.inpw[k * HDIM + t];
            fcol[k] = P.fourier[k * 128 + fc];
        }
        const float bn = P.inpb[t];
        for (int r = 0; r < ROWS; ++r) {
            float a1 = bn, a2 = 0.f;
#pragma unroll
            for (int k = 0; k < DIM; ++k) {
                const float xv = xl[r * DIM + k];
                a1 = __builtin_fmaf(xv, wcol[k], a1);
                a2 = __builtin_fmaf(xv, fcol[k], a2);
            }
            const float ff = (t < 128) ? __sinf(a2) : __cosf(a2);
            const float hv = siluf(a1) + ff;
            h32[r * HDIM + t] = hv;
            h16[r * SH + t]   = (half_t)hv;
        }
    }
    stage_wait(t);   // chunk0 (W1.l0 K0) ready in wb[0]; also publishes stage-0 h

    // ---------- 6 residual blocks: WMMA with TDM double-buffered weights ----------
#pragma unroll 1
    for (int l = 0; l < 6; ++l) {
        const half_t* W1 = P.wt + (size_t)(2 * l)     * 65536;
        const half_t* W2 = P.wt + (size_t)(2 * l + 1) * 65536;
        const float* b1p = P.b1[l];
        const float* b2p = P.b2[l];
        v8f acc[8];

        // Phase 0: GEMM1 K=[0,128) on wb[cur]; DMA W1 K-half 1 into wb[cur^1]
        issue_stage(W1 + 128, wb[cur ^ 1], woff[cur ^ 1], 256, t);
#pragma unroll
        for (int i = 0; i < 8; ++i) acc[i] = (v8f){0.f,0.f,0.f,0.f,0.f,0.f,0.f,0.f};
        gemm_half8(h16, wb[cur], 0, wave, lane, acc);
        stage_wait(t); cur ^= 1;

        // Phase 1: GEMM1 K=[128,256); DMA W2 K-half 0; epilogue silu -> t16
        issue_stage(W2, wb[cur ^ 1], woff[cur ^ 1], 256, t);
        gemm_half8(h16, wb[cur], 128, wave, lane, acc);
        {
            int idx = 0;
#pragma unroll
            for (int mt = 0; mt < 4; ++mt) {
#pragma unroll
                for (int nn = 0; nn < 2; ++nn, ++idx) {
                    const int col = (wave * 2 + nn) * 16 + (lane & 15);
                    const float bb = b1p[col];
                    const int rb = mt * 16 + ((lane >> 4) << 3);
#pragma unroll
                    for (int j = 0; j < 8; ++j)
                        t16[(rb + j) * SH + col] = (half_t)siluf(acc[idx][j] + bb);
                }
            }
        }
        stage_wait(t); cur ^= 1;

        // Phase 2: GEMM2 K=[0,128) (A = t16); DMA W2 K-half 1
        issue_stage(W2 + 128, wb[cur ^ 1], woff[cur ^ 1], 256, t);
#pragma unroll
        for (int i = 0; i < 8; ++i) acc[i] = (v8f){0.f,0.f,0.f,0.f,0.f,0.f,0.f,0.f};
        gemm_half8(t16, wb[cur], 0, wave, lane, acc);
        stage_wait(t); cur ^= 1;

        // Phase 3: GEMM2 K=[128,256); DMA next W1 (or head weights); residual epilogue
        {
            const half_t* nxt = (l < 5) ? (P.wt + (size_t)(2 * l + 2) * 65536) : wh;
            issue_stage(nxt, wb[cur ^ 1], woff[cur ^ 1], (l < 5) ? 256 : 160, t);
        }
        gemm_half8(t16, wb[cur], 128, wave, lane, acc);
        {
            int idx = 0;
#pragma unroll
            for (int mt = 0; mt < 4; ++mt) {
#pragma unroll
                for (int nn = 0; nn < 2; ++nn, ++idx) {
                    const int col = (wave * 2 + nn) * 16 + (lane & 15);
                    const float bb = b2p[col];
                    const int rb = mt * 16 + ((lane >> 4) << 3);
#pragma unroll
                    for (int j = 0; j < 8; ++j) {
                        const float v = h32[(rb + j) * HDIM + col] + acc[idx][j] + bb;
                        h32[(rb + j) * HDIM + col] = v;
                        h16[(rb + j) * SH + col]   = (half_t)v;
                    }
                }
            }
        }
        stage_wait(t); cur ^= 1;
    }

    // ---------- LayerNorm over H=256 ----------
    {
        const int r = t & 63, q = t >> 6;
        float s = 0.f, ss = 0.f;
        const int c0 = q * 64;
        for (int c = 0; c < 64; ++c) {
            const float v = h32[r * HDIM + c0 + c];
            s += v;
            ss = __builtin_fmaf(v, v, ss);
        }
        reds[q * 64 + r] = s;
        redq[q * 64 + r] = ss;
    }
    __syncthreads();
    if (t < 64) {
        const float s4 = reds[t] + reds[64 + t] + reds[128 + t] + reds[192 + t];
        const float q4 = redq[t] + redq[64 + t] + redq[128 + t] + redq[192 + t];
        const float mu = s4 * (1.f / 256.f);
        const float var = q4 * (1.f / 256.f) - mu * mu;
        muv[t] = mu;
        rsv[t] = rsqrtf(var + 1e-5f);
    }
    __syncthreads();
    {
        const float g = P.ln_g[t], be = P.ln_b[t];
        for (int r = 0; r < ROWS; ++r) {
            const float v = (h32[r * HDIM + t] - muv[r]) * rsv[r] * g + be;
            h16[r * SH + t] = (half_t)v;
        }
    }
    __syncthreads();

    // ---------- head layer 1 (em1|th1|st1 = 160 cols), WMMA, K-split pipeline --------
    // wb[cur] already holds head chunk K-half 0 (issued in layer-5 phase 3).
    {
        v8f acc[5];
#pragma unroll
        for (int i = 0; i < 5; ++i) acc[i] = (v8f){0.f,0.f,0.f,0.f,0.f,0.f,0.f,0.f};

        // Phase H0: K=[0,128); DMA head K-half 1
        issue_stage(wh + 128, wb[cur ^ 1], woff[cur ^ 1], 160, t);
#pragma unroll
        for (int i = 0; i < 5; ++i) {
            const int tile = wave + 8 * i;
            const int mt = tile & 3, nt = tile >> 2;
#pragma unroll
            for (int kt = 0; kt < 4; ++kt) {
                v16h a = load_frag(h16, mt * 16, SH, kt * 32, lane);
                v16h b = load_frag(wb[cur], nt * 16, SWC, kt * 32, lane);
                acc[i] = __builtin_amdgcn_wmma_f32_16x16x32_f16(false, a, false, b,
                                                                (short)0, acc[i],
                                                                false, false);
            }
        }
        stage_wait(t); cur ^= 1;

        // Phase H1: K=[128,256); silu epilogue -> t16
#pragma unroll
        for (int i = 0; i < 5; ++i) {
            const int tile = wave + 8 * i;
            const int mt = tile & 3, nt = tile >> 2;
#pragma unroll
            for (int kt = 0; kt < 4; ++kt) {
                v16h a = load_frag(h16, mt * 16, SH, 128 + kt * 32, lane);
                v16h b = load_frag(wb[cur], nt * 16, SWC, kt * 32, lane);
                acc[i] = __builtin_amdgcn_wmma_f32_16x16x32_f16(false, a, false, b,
                                                                (short)0, acc[i],
                                                                false, false);
            }
            const int c = (tile >> 2) * 16 + (lane & 15);
            const float bb = (c < 64) ? P.em1b[c]
                           : (c < 128 ? P.th1b[c - 64] : P.st1b[c - 128]);
            const int rb = (tile & 3) * 16 + ((lane >> 4) << 3);
#pragma unroll
            for (int j = 0; j < 8; ++j)
                t16[(rb + j) * SH + c] = (half_t)siluf(acc[i][j] + bb);
        }
        __syncthreads();
    }

    // ---------- head layer 2 (tiny) + sigmoid + store ----------
    for (int idx = t; idx < ROWS * 7; idx += 256) {
        const int r = idx / 7, o = idx % 7;
        const long gr = (long)row0 + r;
        if (o < 4) {
            float a = P.em2b[o];
#pragma unroll
            for (int k = 0; k < 64; ++k)
                a = __builtin_fmaf((float)t16[r * SH + k], P.em2w[k * 4 + o], a);
            P.out[gr * 4 + o] = sigmoidf_(a);
        } else if (o < 6) {
            const int j = o - 4;
            float a = P.th2b[j];
#pragma unroll
            for (int k = 0; k < 64; ++k)
                a = __builtin_fmaf((float)t16[r * SH + 64 + k], P.th2w[k * 2 + j], a);
            P.out[(long)4 * BTOT + gr * 2 + j] = sigmoidf_(a);
        } else {
            float a = P.st2b[0];
#pragma unroll
            for (int k = 0; k < 32; ++k)
                a = __builtin_fmaf((float)t16[r * SH + 128 + k], P.st2w[k], a);
            P.out[(long)6 * BTOT + gr] = sigmoidf_(a);
        }
    }
}

extern "C" void kernel_launch(void* const* d_in, const int* in_sizes, int n_in,
                              void* d_out, int out_size, void* d_ws, size_t ws_size,
                              hipStream_t stream) {
    // flatten order:
    // 0 x, 1 fourier, 2 inp.w, 3 inp.b,
    // 4+4i blk[i].l1.w, 5+4i blk[i].l1.b, 6+4i blk[i].l2.w, 7+4i blk[i].l2.b (i=0..5),
    // 28 ln_g, 29 ln_b, 30 em1.w, 31 em1.b, 32 em2.w, 33 em2.b,
    // 34 th1.w, 35 th1.b, 36 th2.w, 37 th2.b, 38 st1.w, 39 st1.b, 40 st2.w, 41 st2.b
    (void)in_sizes; (void)n_in; (void)out_size; (void)ws_size;

    half_t* wt = (half_t*)d_ws;             // 12*65536 halves for block weights
    half_t* whd = wt + (size_t)12 * 65536;  // 160*256 halves for head-1 weights

    WPack wp;
    for (int i = 0; i < 6; ++i) {
        wp.w[2 * i]     = (const float*)d_in[4 + 4 * i];   // l1.w
        wp.w[2 * i + 1] = (const float*)d_in[6 + 4 * i];   // l2.w
    }
    prep_block_w<<<3072, 256, 0, stream>>>(wp, wt);
    prep_head_w<<<160, 256, 0, stream>>>((const float*)d_in[30],
                                         (const float*)d_in[34],
                                         (const float*)d_in[38], whd);

    MainArgs P;
    P.x       = (const float*)d_in[0];
    P.fourier = (const float*)d_in[1];
    P.inpw    = (const float*)d_in[2];
    P.inpb    = (const float*)d_in[3];
    for (int i = 0; i < 6; ++i) {
        P.b1[i] = (const float*)d_in[5 + 4 * i];
        P.b2[i] = (const float*)d_in[7 + 4 * i];
    }
    P.ln_g = (const float*)d_in[28];
    P.ln_b = (const float*)d_in[29];
    P.em1b = (const float*)d_in[31];
    P.em2w = (const float*)d_in[32];
    P.em2b = (const float*)d_in[33];
    P.th1b = (const float*)d_in[35];
    P.th2w = (const float*)d_in[36];
    P.th2b = (const float*)d_in[37];
    P.st1b = (const float*)d_in[39];
    P.st2w = (const float*)d_in[40];
    P.st2b = (const float*)d_in[41];
    P.wt   = wt;
    P.out  = (float*)d_out;

    pinn_fused<<<BTOT / ROWS, 256, SMEM_BYTES, stream>>>(P);
}
